// Cross_MHA_90366111908416
// MI455X (gfx1250) — matrix operand order repaired and checked by reading, not compiled, not run on hardware
//
#include <hip/hip_runtime.h>

// ---------------------------------------------------------------------------
// Cross-MHA for MI455X (gfx1250, wave32, WMMA + TDM).
// Pipeline (all bf16 staging in d_ws, f32 accumulate in WMMA):
//   1. convert y,z,Wq,Wk,Wv,Wo f32 -> bf16
//   2. Q = y@Wq+bq, K = z@Wk+bk   (batched over heads, layout [H, B*S, 128])
//   3. per batch b:
//        V   = z_b @ Wv + bv                     [H, 2048, 1024]  bf16
//        P   = softmax(Q K^T / sqrt(128))        [H, 2048, 2048]  bf16 (fused WMMA kernel)
//        Oc  = P @ V  -> concat layout           [2048, 8*1024]   bf16 (ldc trick, no transpose)
//        out_b = Oc @ Wo + bo                    [2048, 1024]     f32
// GEMM A-tiles are staged into LDS by the Tensor Data Mover (tensor_load_to_lds
// with hardware padding reproducing the [128][34]-ushort conflict-free layout),
// fenced with s_wait_tensorcnt + workgroup barrier. Workspace: ~228 MiB.
// ---------------------------------------------------------------------------

#define B_SZ   4
#define H_SZ   8
#define SD     2048
#define SE     2048
#define DKQ    128
#define DMODEL 1024
#define SCALE  0.08838834764831845f   // 1/sqrt(128)

typedef __attribute__((ext_vector_type(16))) __bf16 v16bf;
typedef __attribute__((ext_vector_type(8)))  float  v8f;
typedef __attribute__((ext_vector_type(4)))  unsigned int u32x4;
typedef __attribute__((ext_vector_type(4)))  int          i32x4;
typedef __attribute__((ext_vector_type(8)))  int          i32x8;

union Frag16 { v16bf v; unsigned int u[8]; };

#if defined(__gfx1250__) && __has_builtin(__builtin_amdgcn_tensor_load_to_lds) && \
    __has_builtin(__builtin_amdgcn_s_wait_tensorcnt)
#define USE_TDM 1
#else
#define USE_TDM 0
#endif

__device__ __forceinline__ unsigned short f2bf(float x) {
    unsigned int u = __float_as_uint(x);
    return (unsigned short)((u + 0x7FFFu + ((u >> 16) & 1u)) >> 16);
}

// ------------------------------ f32 -> bf16 --------------------------------
__global__ void cvt_f32_bf16(const float* __restrict__ in,
                             unsigned short* __restrict__ out, long long n) {
    long long i = (long long)blockIdx.x * blockDim.x + threadIdx.x;
    long long stride = (long long)gridDim.x * blockDim.x;
    for (; i < n; i += stride) out[i] = f2bf(in[i]);
}

// ------------------------------ generic GEMM -------------------------------
// C[z] = A[z] @ B[z] + bias[z];  A row-major [M,lda] bf16, B row-major [K,ldb] bf16.
// Block tile 128M x 64N, K-step 32. 8 waves, wave w -> rows [16w,16w+16).
template <bool OUT_BF16>
__global__ __launch_bounds__(256) void gemm_bf16_kernel(
    const unsigned short* __restrict__ A, long long sA,
    const unsigned short* __restrict__ B, long long sB,
    void* __restrict__ C, long long sC,
    const float* __restrict__ bias, long long sBias,
    int K, int lda, int ldb, int ldc) {

    __shared__ unsigned short As[128][34];   // pad to 17 banks (coprime with 64)
    __shared__ unsigned short Bst[64][34];   // B tile stored transposed [n][k]

    const int tid  = threadIdx.x;
    const int wave = tid >> 5;
    const int lane = tid & 31;
    const int l15  = lane & 15;
    const int half = lane >> 4;

    const int n0 = blockIdx.x * 64;
    const int m0 = blockIdx.y * 128;

    const unsigned short* Ab = A + (size_t)blockIdx.z * sA;
    const unsigned short* Bb = B + (size_t)blockIdx.z * sB;
    const float* biasz = bias ? bias + (size_t)blockIdx.z * sBias : nullptr;

    v8f acc[4] = {};

    const int bk = tid >> 3, bj = (tid & 7) * 8;    // B fill: k row, 8-elem chunk

#if USE_TDM
    // TDM descriptor (2D tile 32x128 elems of 2B, LDS rows padded 16DW + 1DW).
    // Static part of group 1 (CDNA5 ISA).
    const unsigned d0 = (unsigned)lda;              // tensor_dim0 (OOB bound)
    const unsigned d1 = 1u << 30;                   // tensor_dim1 (never OOB here)
    i32x8 g1;
    g1[0] = (int)((1u << 16) |                      // data_size   = 1 (2 bytes)
                  (1u << 20) |                      // pad_enable
                  (3u << 22));                      // pad_interval = 3 (16 DWORDs)
                                                    // pad_amount   = 0 (1 DWORD)
    g1[1] = (int)((d0 & 0xFFFFu) << 16);            // tensor_dim0[15:0]
    g1[2] = (int)((d0 >> 16) | ((d1 & 0xFFFFu) << 16)); // dim0[31:16] | dim1[15:0]
    g1[3] = (int)((d1 >> 16) | (32u << 16));        // dim1[31:16] | tile_dim0 = 32
    g1[4] = 128;                                    // tile_dim1 = 128 rows
    g1[5] = (int)(unsigned)lda;                     // tensor_dim0_stride[31:0]
    g1[6] = 0;                                      // stride hi, dim1_stride lo
    g1[7] = 0;
    const i32x4 gz4 = {0, 0, 0, 0};
#if __clang_major__ >= 23
    const i32x8 gz8 = {0, 0, 0, 0, 0, 0, 0, 0};
#endif
    const unsigned lds_base = (unsigned)(size_t)(&As[0][0]);
    const unsigned long long ga_row0 =
        (unsigned long long)(size_t)(Ab + (size_t)m0 * lda);
#else
    const int ar = tid >> 1, ah = tid & 1;          // A fill: row, 16-elem half
#endif

    for (int k0 = 0; k0 < K; k0 += 32) {
#if USE_TDM
        // ---- stage A tile 128x32 via Tensor Data Mover (wave 0 issues) ----
        if (wave == 0) {
            const unsigned long long ga = ga_row0 + (unsigned long long)k0 * 2ull;
            u32x4 g0;
            g0[0] = 1u;                                       // count = 1 (valid)
            g0[1] = lds_base;                                 // lds_addr
            g0[2] = (unsigned)(ga & 0xFFFFFFFFull);           // global_addr[31:0]
            g0[3] = (unsigned)((ga >> 32) & 0x01FFFFFFull)    // global_addr[56:32]
                  | (2u << 30);                               // type = 2 (image)
#if __clang_major__ >= 23
            __builtin_amdgcn_tensor_load_to_lds(g0, g1, gz4, gz4, gz8, 0);
#else
            __builtin_amdgcn_tensor_load_to_lds(g0, g1, gz4, gz4, 0);
#endif
        }
#else
        // ---- stage A tile 128x32 (manual path) ----
        uint4 av = *(const uint4*)(Ab + (size_t)(m0 + ar) * lda + k0 + ah * 16);
        unsigned int* adst = (unsigned int*)&As[ar][ah * 16];
        adst[0] = av.x; adst[1] = av.y; adst[2] = av.z; adst[3] = av.w;
#endif
        // ---- stage B tile 32x64, transposed ----
        union { uint4 q; unsigned short s[8]; } bv;
        bv.q = *(const uint4*)(Bb + (size_t)(k0 + bk) * ldb + n0 + bj);
#pragma unroll
        for (int e = 0; e < 8; ++e) Bst[bj + e][bk] = bv.s[e];
#if USE_TDM
        if (wave == 0) __builtin_amdgcn_s_wait_tensorcnt(0);
#endif
        __syncthreads();

        // ---- A fragment (16x32), ISA layout ----
        Frag16 af;
        const int kbA = half * 8;
#pragma unroll
        for (int v = 0; v < 8; ++v) {
            int k = (v < 4) ? (kbA + 2 * v) : (16 + kbA + 2 * (v - 4));
            af.u[v] = *(const unsigned int*)&As[wave * 16 + l15][k];
        }
        // ---- 4 B fragments + WMMA ----
#pragma unroll
        for (int nt = 0; nt < 4; ++nt) {
            Frag16 bf;
#pragma unroll
            for (int v = 0; v < 8; ++v)
                bf.u[v] = *(const unsigned int*)&Bst[nt * 16 + l15][(v < 4) ? (half * 8 + 2 * v) : (16 + half * 8 + 2 * (v - 4))];
            acc[nt] = __builtin_amdgcn_wmma_f32_16x16x32_bf16(
                false, af.v, false, bf.v, (short)0, acc[nt], false, false);
        }
        __syncthreads();
    }

    // ---- epilogue: C layout (VGPR r -> m = r + 8*half), lane n = l15 ----
#pragma unroll
    for (int nt = 0; nt < 4; ++nt) {
        const int n = n0 + nt * 16 + l15;
        const float bval = biasz ? biasz[n] : 0.0f;
#pragma unroll
        for (int r = 0; r < 8; ++r) {
            const int m = m0 + wave * 16 + half * 8 + r;
            const float val = acc[nt][r] + bval;
            if constexpr (OUT_BF16) {
                unsigned short* Cb = (unsigned short*)C + (size_t)blockIdx.z * sC;
                Cb[(size_t)m * ldc + n] = f2bf(val);
            } else {
                float* Cb = (float*)C + (size_t)blockIdx.z * sC;
                Cb[(size_t)m * ldc + n] = val;
            }
        }
    }
}

// --------------------- fused scores + softmax kernel -----------------------
// Block = (16 dec rows) x (head). 8 waves split 2048 enc cols (256 each).
// Q layout [H, B, SD, 128], K layout [H, B, SE, 128], P layout [H, SD, SE].
__global__ __launch_bounds__(256) void scores_softmax_kernel(
    const unsigned short* __restrict__ Q, const unsigned short* __restrict__ Km,
    unsigned short* __restrict__ P, int b) {

    __shared__ float redmax[8][16];
    __shared__ float redsum[8][16];

    const int h  = blockIdx.y;
    const int s0 = blockIdx.x * 16;
    const int tid  = threadIdx.x;
    const int wave = tid >> 5;
    const int lane = tid & 31;
    const int l15  = lane & 15;
    const int half = lane >> 4;

    const unsigned short* Qh = Q + ((size_t)(h * B_SZ + b) * SD + s0) * DKQ;
    const unsigned short* Kh = Km + (size_t)(h * B_SZ + b) * SE * DKQ;
    unsigned short* Ph = P + (size_t)h * SD * SE;

    // Q fragments: 16 rows x 128 -> 4 A-fragments, loaded once from global.
    Frag16 aq[4];
    const int kbA = half * 8;
#pragma unroll
    for (int c = 0; c < 4; ++c)
#pragma unroll
        for (int v = 0; v < 8; ++v) {
            int k = c * 32 + ((v < 4) ? (kbA + 2 * v) : (16 + kbA + 2 * (v - 4)));
            aq[c].u[v] = *(const unsigned int*)(Qh + (size_t)l15 * DKQ + k);
        }

    // scores for this wave's 256 columns: 16 D-tiles of 16x16 f32
    v8f sc[16];
    const int colbase = wave * 256;
#pragma unroll
    for (int nt = 0; nt < 16; ++nt) {
        v8f acc = {};
        const unsigned short* Kn = Kh + (size_t)(colbase + nt * 16 + l15) * DKQ;
#pragma unroll
        for (int c = 0; c < 4; ++c) {
            Frag16 bf;
#pragma unroll
            for (int v = 0; v < 8; ++v)
                bf.u[v] = *(const unsigned int*)(Kn + c * 32 + ((v < 4) ? (half * 8 + 2 * v) : (16 + half * 8 + 2 * (v - 4))));
            acc = __builtin_amdgcn_wmma_f32_16x16x32_bf16(
                false, aq[c].v, false, bf.v, (short)0, acc, false, false);
        }
        sc[nt] = acc * SCALE;
    }

    // ---- row max: local over 16 tiles, shfl within 16-lane half, LDS across waves
    float rmax[8];
#pragma unroll
    for (int r = 0; r < 8; ++r) {
        float m_ = -3.0e38f;
#pragma unroll
        for (int nt = 0; nt < 16; ++nt) m_ = fmaxf(m_, sc[nt][r]);
        m_ = fmaxf(m_, __shfl_xor(m_, 1));
        m_ = fmaxf(m_, __shfl_xor(m_, 2));
        m_ = fmaxf(m_, __shfl_xor(m_, 4));
        m_ = fmaxf(m_, __shfl_xor(m_, 8));
        rmax[r] = m_;
    }
    if (l15 == 0)
#pragma unroll
        for (int r = 0; r < 8; ++r) redmax[wave][half * 8 + r] = rmax[r];
    __syncthreads();
#pragma unroll
    for (int r = 0; r < 8; ++r) {
        float m_ = -3.0e38f;
#pragma unroll
        for (int w = 0; w < 8; ++w) m_ = fmaxf(m_, redmax[w][half * 8 + r]);
        rmax[r] = m_;
    }

    // ---- exp + row sum ----
    float rsum[8];
#pragma unroll
    for (int r = 0; r < 8; ++r) {
        float s_ = 0.0f;
        const float mx = rmax[r];
#pragma unroll
        for (int nt = 0; nt < 16; ++nt) {
            float e = __expf(sc[nt][r] - mx);
            sc[nt][r] = e;
            s_ += e;
        }
        s_ += __shfl_xor(s_, 1);
        s_ += __shfl_xor(s_, 2);
        s_ += __shfl_xor(s_, 4);
        s_ += __shfl_xor(s_, 8);
        rsum[r] = s_;
    }
    if (l15 == 0)
#pragma unroll
        for (int r = 0; r < 8; ++r) redsum[wave][half * 8 + r] = rsum[r];
    __syncthreads();
#pragma unroll
    for (int r = 0; r < 8; ++r) {
        float s_ = 0.0f;
#pragma unroll
        for (int w = 0; w < 8; ++w) s_ += redsum[w][half * 8 + r];
        rsum[r] = 1.0f / s_;
    }

    // ---- normalize + store bf16 P ----
#pragma unroll
    for (int nt = 0; nt < 16; ++nt) {
        const int col = colbase + nt * 16 + l15;
#pragma unroll
        for (int r = 0; r < 8; ++r) {
            const int row = s0 + half * 8 + r;
            Ph[(size_t)row * SE + col] = f2bf(sc[nt][r] * rsum[r]);
        }
    }
}

// ------------------------------- launcher ----------------------------------
extern "C" void kernel_launch(void* const* d_in, const int* in_sizes, int n_in,
                              void* d_out, int out_size, void* d_ws, size_t ws_size,
                              hipStream_t stream) {
    (void)in_sizes; (void)n_in; (void)out_size; (void)ws_size;

    const float* z  = (const float*)d_in[0];
    const float* y  = (const float*)d_in[1];
    const float* Wq = (const float*)d_in[2];
    const float* bq = (const float*)d_in[3];
    const float* Wk = (const float*)d_in[4];
    const float* bk = (const float*)d_in[5];
    const float* Wv = (const float*)d_in[6];
    const float* bv = (const float*)d_in[7];
    const float* Wo = (const float*)d_in[8];
    const float* bo = (const float*)d_in[9];

    // workspace layout (ushort elements, all 16B aligned)
    unsigned short* ws = (unsigned short*)d_ws;
    size_t off = 0;
    auto take = [&](size_t elems) {
        unsigned short* p = ws + off;
        off += (elems + 7) & ~(size_t)7;
        return p;
    };
    unsigned short* yb   = take((size_t)B_SZ * SD * DMODEL);          // 8.4M
    unsigned short* zb   = take((size_t)B_SZ * SE * DMODEL);          // 8.4M
    unsigned short* wqb  = take((size_t)H_SZ * DMODEL * DKQ);         // 1.0M
    unsigned short* wkb  = take((size_t)H_SZ * DMODEL * DKQ);         // 1.0M
    unsigned short* wvb  = take((size_t)H_SZ * DMODEL * DMODEL);      // 8.4M
    unsigned short* wob  = take((size_t)H_SZ * DMODEL * DMODEL);      // 8.4M
    unsigned short* qb   = take((size_t)H_SZ * B_SZ * SD * DKQ);      // 8.4M
    unsigned short* kbuf = take((size_t)H_SZ * B_SZ * SE * DKQ);      // 8.4M
    unsigned short* vbuf = take((size_t)H_SZ * SE * DMODEL);          // 16.8M
    unsigned short* pbuf = take((size_t)H_SZ * SD * SE);              // 33.6M
    unsigned short* ocat = take((size_t)SD * H_SZ * DMODEL);          // 16.8M

    // 1. conversions
    auto cvt = [&](const float* src, unsigned short* dst, long long n) {
        cvt_f32_bf16<<<4096, 256, 0, stream>>>(src, dst, n);
    };
    cvt(y,  yb,  (long long)B_SZ * SD * DMODEL);
    cvt(z,  zb,  (long long)B_SZ * SE * DMODEL);
    cvt(Wq, wqb, (long long)H_SZ * DMODEL * DKQ);
    cvt(Wk, wkb, (long long)H_SZ * DMODEL * DKQ);
    cvt(Wv, wvb, (long long)H_SZ * DMODEL * DMODEL);
    cvt(Wo, wob, (long long)H_SZ * DMODEL * DMODEL);

    // 2. Q/K projections, batched over heads: M = B*SD = 8192, N = 128, K = 1024
    {
        dim3 grid(DKQ / 64, (B_SZ * SD) / 128, H_SZ);
        gemm_bf16_kernel<true><<<grid, 256, 0, stream>>>(
            yb, 0LL, wqb, (long long)DMODEL * DKQ,
            (void*)qb, (long long)B_SZ * SD * DKQ, bq, (long long)DKQ,
            DMODEL, DMODEL, DKQ, DKQ);
        gemm_bf16_kernel<true><<<grid, 256, 0, stream>>>(
            zb, 0LL, wkb, (long long)DMODEL * DKQ,
            (void*)kbuf, (long long)B_SZ * SE * DKQ, bk, (long long)DKQ,
            DMODEL, DMODEL, DKQ, DKQ);
    }

    // 3. per-batch attention
    for (int b = 0; b < B_SZ; ++b) {
        // V = z_b @ Wv + bv : M=2048, N=1024, K=1024, batched over h
        {
            dim3 grid(DMODEL / 64, SE / 128, H_SZ);
            gemm_bf16_kernel<true><<<grid, 256, 0, stream>>>(
                zb + (size_t)b * SE * DMODEL, 0LL,
                wvb, (long long)DMODEL * DMODEL,
                (void*)vbuf, (long long)SE * DMODEL, bv, (long long)DMODEL,
                DMODEL, DMODEL, DMODEL, DMODEL);
        }
        // P = softmax(Q K^T * scale)
        {
            dim3 grid(SD / 16, H_SZ);
            scores_softmax_kernel<<<grid, 256, 0, stream>>>(qb, kbuf, pbuf, b);
        }
        // Ocat[:, h*1024:(h+1)*1024] = P_h @ V_h : M=2048, N=1024, K=2048
        {
            dim3 grid(DMODEL / 64, SD / 128, H_SZ);
            gemm_bf16_kernel<true><<<grid, 256, 0, stream>>>(
                pbuf, (long long)SD * SE,
                vbuf, (long long)SE * DMODEL,
                (void*)ocat, (long long)DMODEL, nullptr, 0LL,
                SE, SE, DMODEL, H_SZ * DMODEL);
        }
        // out_b = Ocat @ Wo + bo : M=2048, N=1024, K=8192, f32 out
        {
            dim3 grid(DMODEL / 64, SD / 128, 1);
            gemm_bf16_kernel<false><<<grid, 256, 0, stream>>>(
                ocat, 0LL, wob, 0LL,
                (void*)((float*)d_out + (size_t)b * SD * DMODEL), 0LL,
                bo, 0LL,
                H_SZ * DMODEL, H_SZ * DMODEL, DMODEL, DMODEL);
        }
    }
}
